// Fourier_12317966205391
// MI455X (gfx1250) — matrix operand order repaired and checked by reading, not compiled, not run on hardware
//
#include <hip/hip_runtime.h>

typedef _Float16 v16h __attribute__((ext_vector_type(16)));
typedef _Float16 v8h  __attribute__((ext_vector_type(8)));
typedef float    v8f  __attribute__((ext_vector_type(8)));

#define BATCH 32768
#define IDIM  64
#define UNITS 64
#define NW    65
#define KK    64          // 2*FREQUENCIES (cos/sin interleaved) per input dim
#define M_TILE 128        // rows per workgroup (8 M-subtiles of 16)
#define LDS_STRIDE 520    // 512 halfs + 8 pad (1040 B row stride -> conflict-free b128)

// --- prologue 1: convert w[u][d][1+kk] (f32) -> wh[d][u][kk] (f16) ---
__global__ __launch_bounds__(256) void fourier_wcvt(const float* __restrict__ w,
                                                    _Float16* __restrict__ wh) {
  int idx = blockIdx.x * 256 + threadIdx.x;     // 0 .. 64*64*64-1
  int d  = idx >> 12;
  int u  = (idx >> 6) & 63;
  int kk = idx & 63;
  wh[idx] = (_Float16)w[(u * IDIM + d) * NW + 1 + kk];
}

// --- prologue 2: bias[u] = 0.5 * sum_d w[u][d][0] ---
__global__ void fourier_bias(const float* __restrict__ w, float* __restrict__ bias) {
  int u = threadIdx.x;
  float s = 0.f;
  for (int d = 0; d < IDIM; ++d) s += w[(u * IDIM + d) * NW];
  bias[u] = 0.5f * s;
}

// --- main: [32768 x 4096] (basis, built on the fly) x [4096 x 64] (wh) ---
__global__ __launch_bounds__(256) void fourier_main(const float* __restrict__ x,
                                                    const _Float16* __restrict__ wh,
                                                    const float* __restrict__ bias,
                                                    float* __restrict__ out) {
  __shared__ __align__(16) _Float16 sA[M_TILE * LDS_STRIDE];   // 133 KB

  const int tid    = threadIdx.x;
  const int lane   = tid & 31;
  const int wave   = tid >> 5;      // 8 waves
  const int lane15 = lane & 15;
  const int hi     = (lane >> 4) & 1;
  const int n_sub  = wave & 3;      // 0..3  (N subtile)
  const int m_half = wave >> 2;     // 0..1  (which 4 M-subtiles this wave owns)
  const int u0     = n_sub * 16;
  const int b0     = blockIdx.x * M_TILE;

  // cooperative-basis assignment: 4 (row, d) pairs per thread per super-chunk
  const int c_row = tid >> 3;       // 0..31  (+ p*32)
  const int c_dl  = tid & 7;        // 0..7

  v8f acc[4] = {};

  for (int g = 0; g < 8; ++g) {     // 8 super-chunks of 8 input dims each
    // ---- phase 1: basis -> LDS via sincos + angle-addition recurrence ----
#pragma unroll
    for (int p = 0; p < 4; ++p) {
      const int   row = p * 32 + c_row;           // 0..127
      const int   d   = g * 8 + c_dl;
      const float xv  = x[(b0 + row) * IDIM + d];
      const float th  = 1.57079632679489662f * xv;  // pi/LENGTH, LENGTH = 2
      float s1, c1;
      __sincosf(th, &s1, &c1);
      float c = c1, s = s1;
      _Float16* dst = &sA[row * LDS_STRIDE + c_dl * KK];
#pragma unroll
      for (int grp = 0; grp < 8; ++grp) {
        v8h buf;
#pragma unroll
        for (int j = 0; j < 4; ++j) {
          buf[2 * j]     = (_Float16)c;     // cos_f  -> k = 2(f-1)
          buf[2 * j + 1] = (_Float16)s;     // sin_f  -> k = 2(f-1)+1
          float cn = c * c1 - s * s1;
          float sn = s * c1 + c * s1;
          c = cn; s = sn;
        }
        *(v8h*)(dst + grp * 8) = buf;       // ds_store_b128
      }
    }
    __syncthreads();

    // ---- phase 2: 16 K-chunks x 4 M-subtiles; B fragment reused 4x ----
#pragma unroll
    for (int dl = 0; dl < 8; ++dl) {
      const int d = g * 8 + dl;
      const _Float16* bRow = wh + (size_t)(d * UNITS + u0 + lane15) * KK;
#pragma unroll
      for (int kc = 0; kc < 2; ++kc) {
        // B (32x16, this code's k order): lane n holds 16 contiguous K (+16 for hi)
        union { v16h v; v8h h[2]; } b;
        const _Float16* bp = bRow + kc * 32 + hi * 16;
        b.h[0] = *(const v8h*)bp;           // global_load_b128 (L2-resident)
        b.h[1] = *(const v8h*)(bp + 8);
#pragma unroll
        for (int m = 0; m < 4; ++m) {
          const int msub = m_half * 4 + m;
          // A: lane m holds 16 contiguous K (+16 for hi), matching B
          union { v16h v; v8h h[2]; } a;
          const _Float16* ap = &sA[(msub * 16 + lane15) * LDS_STRIDE
                                   + dl * KK + kc * 32 + hi * 16];
          a.h[0] = *(const v8h*)ap;         // ds_load_b128
          a.h[1] = *(const v8h*)(ap + 8);   // ds_load_b128
          acc[m] = __builtin_amdgcn_wmma_f32_16x16x32_f16(
              false, a.v, false, b.v, (short)0, acc[m], false, false);
        }
      }
    }
    __syncthreads();
  }

  // ---- epilogue: add const-term bias, store C (VGPR j -> row j + 8*hi) ----
  const float bv  = bias[u0 + lane15];
  const int   col = u0 + lane15;
#pragma unroll
  for (int m = 0; m < 4; ++m) {
    const int row0 = b0 + (m_half * 4 + m) * 16 + hi * 8;
#pragma unroll
    for (int j = 0; j < 8; ++j) {
      out[(row0 + j) * UNITS + col] = acc[m][j] + bv;
    }
  }
}

extern "C" void kernel_launch(void* const* d_in, const int* in_sizes, int n_in,
                              void* d_out, int out_size, void* d_ws, size_t ws_size,
                              hipStream_t stream) {
  (void)in_sizes; (void)n_in; (void)out_size; (void)ws_size;
  const float* x = (const float*)d_in[0];
  const float* w = (const float*)d_in[1];

  _Float16* wh   = (_Float16*)d_ws;                                   // 512 KB
  float*    bias = (float*)((char*)d_ws + (size_t)IDIM * UNITS * KK * sizeof(_Float16));

  fourier_wcvt<<<(IDIM * UNITS * KK) / 256, 256, 0, stream>>>(w, wh);
  fourier_bias<<<1, UNITS, 0, stream>>>(w, bias);
  fourier_main<<<BATCH / M_TILE, 256, 0, stream>>>(x, wh, bias, (float*)d_out);
}
